// BertCrf_10892037062973
// MI455X (gfx1250) — matrix operand order repaired and checked by reading, not compiled, not run on hardware
//
#include <hip/hip_runtime.h>
#include <hip/hip_bf16.h>

#define B_ 256
#define S_ 512
#define L_ 128

#define LOG2E 1.4426950408889634f
#define LN2   0.6931471805599453f

typedef __attribute__((ext_vector_type(16))) __bf16 v16bf;
typedef __attribute__((ext_vector_type(8)))  float  v8f;

union BfVec {
  unsigned int u[8];
  v16bf v;
};

// Packed f32 -> bf16 (RNE) pair; lowers to v_cvt_pk_bf16_f32 when available.
__device__ __forceinline__ unsigned int pack2bf(float lo, float hi) {
  float2 v = make_float2(lo, hi);
  __hip_bfloat162 h = __float22bfloat162_rn(v);
  unsigned int r;
  __builtin_memcpy(&r, &h, sizeof(r));
  return r;
}

// Raw hardware transcendentals; all inputs here are provably in the safe
// range (exp2 args bounded by row spread, log args in [~1e-6, 1e8]),
// so no libm range-fixup code is needed.
__device__ __forceinline__ float fast_exp(float x) {
  return __builtin_amdgcn_exp2f(x * LOG2E);
}
__device__ __forceinline__ float fast_log(float x) {
  return __builtin_amdgcn_logf(x) * LN2;
}

// ---------------------------------------------------------------------------
// Kernel 1: CRF forward recursion (log-denominator) via exp-space WMMA GEMMs.
// Grid: B/16 blocks of 256 threads (8 waves). Each block owns 16 batch rows;
// each wave owns a 16x16 tile (16 tag-columns), with its slice of exp(T)
// resident in VGPRs for the whole time loop. Per step: 2 barriers; shift is
// alpha[row][0] (valid: bounded row spread); mask select is a branch-free
// arithmetic blend; all 8 A-operand ds_load_b128s are issued before the
// WMMA chain so dependency waits stagger.
// ---------------------------------------------------------------------------
__global__ __launch_bounds__(256) void crf_forward_kernel(
    const float* __restrict__ feats, const float* __restrict__ startT,
    const float* __restrict__ endT, const float* __restrict__ T,
    const int* __restrict__ mask, float* __restrict__ log_den) {
  __shared__ __align__(16) unsigned short expTt[L_ * L_];   // [n][k] bf16 exp(T[k][n])
  __shared__ __align__(16) float          alphaS[16 * L_];  // running alpha, f32
  __shared__ __align__(16) unsigned short pA[16 * L_];      // bf16 exp(alpha - m)
  __shared__ __align__(16) float          mrow[16];         // per-row shift
  __shared__ unsigned int  maskBits[S_];                    // bit r = mask[b0+r][t]
  __shared__ float         pred[256];                       // final reduction only

  const int tid  = threadIdx.x;
  const int b0   = blockIdx.x * 16;
  const int w    = tid >> 5;
  const int lane = tid & 31;

  // --- one-time staging: exp(T) transposed (bf16), alpha_0, packed mask ---
  for (int i = tid; i < (L_ * L_) / 2; i += 256) {
    int idx = i * 2;
    int n = idx >> 7, k = idx & (L_ - 1);
    *(unsigned int*)&expTt[idx] =
        pack2bf(fast_exp(T[k * L_ + n]), fast_exp(T[(k + 1) * L_ + n]));
  }
  for (int i = tid; i < 16 * L_; i += 256) {
    int r = i >> 7, j = i & (L_ - 1);
    alphaS[i] = startT[j] + feats[(size_t)(b0 + r) * S_ * L_ + j];
  }
  for (int t = tid; t < S_; t += 256) {
    unsigned int mb = 0;
#pragma unroll
    for (int r = 0; r < 16; ++r)
      mb |= (mask[(b0 + r) * S_ + t] != 0 ? 1u : 0u) << r;
    maskBits[t] = mb;
  }
  __syncthreads();

  // --- preload this wave's B operand (128x16 slice of exp(T)) into VGPRs ---
  // B k order used here (32x16 per WMMA): lanes 0-15 -> K=0..15 (N=lane),
  // lanes 16-31 -> K=16..31 (N=lane-16); VGPR v holds K = off+2v, off+2v+1.
  BfVec breg[4];
  {
    const int Nn0   = (w << 4) + (lane & 15);
    const int koffB = (lane >> 4) << 4;
#pragma unroll
    for (int c = 0; c < 4; ++c)
#pragma unroll
      for (int v = 0; v < 8; ++v)
        breg[c].u[v] =
            *(const unsigned int*)&expTt[Nn0 * L_ + c * 32 + koffB + v * 2];
  }

  const int rrow  = tid >> 4;                // panel row (0..15)
  const int seg   = tid & 15;                // panel segment of 8 elems
  const int Mi    = lane & 15;               // A-matrix row
  const int koffA = ((lane >> 4) & 1) * 8;
  const int Nn    = (w << 4) + (lane & 15);  // output column
  const int Mbase = (lane >> 4) << 3;        // output row base (C layout)

  // This thread's 8 alpha values (rows Mbase..Mbase+7, column Nn) in regs.
  float aval[8];
#pragma unroll
  for (int r2 = 0; r2 < 8; ++r2) aval[r2] = alphaS[(Mbase + r2) * L_ + Nn];

  for (int t = 1; t < S_; ++t) {
    // ---- phase 1: bf16 exp panel, shift = alpha[row][0] (no reduction) ----
    {
      if (tid < 16) mrow[tid] = alphaS[tid * L_];
      float m = alphaS[rrow * L_];
      const float4 a0 = *(const float4*)&alphaS[rrow * L_ + seg * 8];
      const float4 a1 = *(const float4*)&alphaS[rrow * L_ + seg * 8 + 4];
      uint4 pk;
      pk.x = pack2bf(fast_exp(a0.x - m), fast_exp(a0.y - m));
      pk.y = pack2bf(fast_exp(a0.z - m), fast_exp(a0.w - m));
      pk.z = pack2bf(fast_exp(a1.x - m), fast_exp(a1.y - m));
      pk.w = pack2bf(fast_exp(a1.z - m), fast_exp(a1.w - m));
      *(uint4*)&pA[rrow * L_ + seg * 8] = pk;
    }
    __syncthreads();

    // ---- phase 2: 16x128 @ 128x16 GEMM in exp space (4 bf16 WMMAs) ----
    // Issue all 8 A-operand ds_load_b128s first so waits stagger.
    BfVec aops[4];
#pragma unroll
    for (int c = 0; c < 4; ++c)
#pragma unroll
      for (int v = 0; v < 8; ++v) {
        // A k order used here (16x32): lanes 0-15: K=0..15; lanes 16-31: K=16..31
        // (matches B). VGPR v packs K, K+1 -> contiguous b32 in pA.
        int K = c * 32 + koffA * 2 + v * 2;
        aops[c].u[v] = *(const unsigned int*)&pA[Mi * L_ + K];
      }
    v8f acc = {0.f, 0.f, 0.f, 0.f, 0.f, 0.f, 0.f, 0.f};
#pragma unroll
    for (int c = 0; c < 4; ++c)
      acc = __builtin_amdgcn_wmma_f32_16x16x32_bf16(
          false, aops[c].v, false, breg[c].v, (short)0, acc, false, false);

    // ---- epilogue: batched feat loads, log + shift + emission, blend ----
    float fv[8];
    const float* fbase = &feats[((size_t)(b0 + Mbase) * S_ + t) * L_ + Nn];
#pragma unroll
    for (int r2 = 0; r2 < 8; ++r2) {
      fv[r2] = fbase[(size_t)r2 * S_ * L_];
      __builtin_prefetch(fbase + (size_t)r2 * S_ * L_ + L_, 0, 1);
    }
    const unsigned int mb = maskBits[t];
    const float4 mr0 = *(const float4*)&mrow[Mbase];
    const float4 mr1 = *(const float4*)&mrow[Mbase + 4];
    const float mr[8] = {mr0.x, mr0.y, mr0.z, mr0.w, mr1.x, mr1.y, mr1.z, mr1.w};
#pragma unroll
    for (int r2 = 0; r2 < 8; ++r2) {
      float nv  = fast_log(acc[r2]) + mr[r2] + fv[r2];
      // Branch-free blend: sel in {0,1}; nv is always computed & consumed,
      // so the compiler cannot sink the loads/logs into divergent branches.
      float sel = (float)((mb >> (Mbase + r2)) & 1u);
      aval[r2]  = fmaf(sel, nv - aval[r2], aval[r2]);
      alphaS[(Mbase + r2) * L_ + Nn] = aval[r2];
    }
    __syncthreads();
  }

  // --- final logsumexp_j(alpha + endT) per batch row ---
  {
    float mx = -3.4e38f;
#pragma unroll
    for (int q = 0; q < 8; ++q) {
      int j = seg * 8 + q;
      mx = fmaxf(mx, alphaS[rrow * L_ + j] + endT[j]);
    }
    pred[tid] = mx;
    __syncthreads();
    if (tid < 16) {
      float m = -3.4e38f;
#pragma unroll
      for (int q = 0; q < 16; ++q) m = fmaxf(m, pred[tid * 16 + q]);
      mrow[tid] = m;
    }
    __syncthreads();
    float sm = 0.f;
#pragma unroll
    for (int q = 0; q < 8; ++q) {
      int j = seg * 8 + q;
      sm += fast_exp(alphaS[rrow * L_ + j] + endT[j] - mrow[rrow]);
    }
    pred[tid] = sm;
    __syncthreads();
    if (tid < 16) {
      float s = 0.f;
#pragma unroll
      for (int q = 0; q < 16; ++q) s += pred[tid * 16 + q];
      log_den[b0 + tid] = fast_log(s) + mrow[tid];
    }
  }
}

// ---------------------------------------------------------------------------
// Kernel 2: numerator path score per batch (gathers + reduction).
// ---------------------------------------------------------------------------
__global__ __launch_bounds__(256) void crf_numerator_kernel(
    const float* __restrict__ feats, const float* __restrict__ startT,
    const float* __restrict__ endT, const float* __restrict__ T,
    const int* __restrict__ mask, const int* __restrict__ labels,
    float* __restrict__ log_num) {
  __shared__ float sred[256];
  __shared__ int   mred[256];
  const int b = blockIdx.x;
  const int tid = threadIdx.x;
  float part = 0.f;
  int   msum = 0;
  for (int t = tid; t < S_; t += 256) {
    int lb = labels[b * S_ + t]; if (lb < 0) lb = 0;
    int mt = mask[b * S_ + t];
    msum += mt;
    if (t == 0) {
      part += startT[lb] + feats[(size_t)b * S_ * L_ + lb];
    } else {
      int lp = labels[b * S_ + t - 1]; if (lp < 0) lp = 0;
      float c = T[lp * L_ + lb] + feats[((size_t)b * S_ + t) * L_ + lb];
      part += mt ? c : 0.f;
    }
  }
  sred[tid] = part;
  mred[tid] = msum;
  __syncthreads();
  for (int off = 128; off > 0; off >>= 1) {
    if (tid < off) { sred[tid] += sred[tid + off]; mred[tid] += mred[tid + off]; }
    __syncthreads();
  }
  if (tid == 0) {
    int slen = mred[0] - 1;
    int lt = labels[b * S_ + slen]; if (lt < 0) lt = 0;
    log_num[b] = sred[0] + endT[lt];
  }
}

// ---------------------------------------------------------------------------
// Kernel 3: mean((log_den - log_num) * confidence)
// ---------------------------------------------------------------------------
__global__ __launch_bounds__(256) void crf_finalize_kernel(
    const float* __restrict__ log_den, const float* __restrict__ log_num,
    const float* __restrict__ conf, float* __restrict__ out) {
  __shared__ float sred[256];
  int tid = threadIdx.x;
  sred[tid] = (log_den[tid] - log_num[tid]) * conf[tid];
  __syncthreads();
  for (int off = 128; off > 0; off >>= 1) {
    if (tid < off) sred[tid] += sred[tid + off];
    __syncthreads();
  }
  if (tid == 0) out[0] = sred[0] * (1.0f / (float)B_);
}

extern "C" void kernel_launch(void* const* d_in, const int* in_sizes, int n_in,
                              void* d_out, int out_size, void* d_ws, size_t ws_size,
                              hipStream_t stream) {
  const float* feats  = (const float*)d_in[0];
  const float* startT = (const float*)d_in[1];
  const float* endT   = (const float*)d_in[2];
  const float* T      = (const float*)d_in[3];
  const float* conf   = (const float*)d_in[4];
  const int*   mask   = (const int*)d_in[5];
  const int*   labels = (const int*)d_in[6];

  float* log_den = (float*)d_ws;
  float* log_num = log_den + B_;

  crf_forward_kernel<<<B_ / 16, 256, 0, stream>>>(feats, startT, endT, T, mask, log_den);
  crf_numerator_kernel<<<B_, 256, 0, stream>>>(feats, startT, endT, T, mask, labels, log_num);
  crf_finalize_kernel<<<1, 256, 0, stream>>>(log_den, log_num, conf, (float*)d_out);
}
